// SAModule_26594437497541
// MI455X (gfx1250) — compile-verified, matrix operand order checked by reading, not run on hardware
//
#include <hip/hip_runtime.h>
#include <stdint.h>

#define NPTS 8192
#define MPTS 2048
#define BEV  4
#define KNN  64
#define DIN  32
#define R2C  0.04f
#define BIGF 1.0e10f

typedef __attribute__((ext_vector_type(8)))  __bf16 bf16x8;
typedef __attribute__((ext_vector_type(16))) __bf16 bf16x16;
typedef __attribute__((ext_vector_type(8)))  float  f32x8;

// ---------------------------------------------------------------------------
// Kernel 1: farthest point sampling. One block per event, 1024 threads.
// Each thread owns 8 points (coords cached in registers) -> 2 barriers/iter.
// ---------------------------------------------------------------------------
__global__ void fps_kernel(const float* __restrict__ pos, float* __restrict__ centers) {
  const int b    = blockIdx.x;
  const int tid  = threadIdx.x;
  const int lane = tid & 31;
  const int wv   = tid >> 5;
  const float* p = pos + (size_t)b * NPTS * 3;

  __shared__ float dsh[NPTS];
  __shared__ float rv[32];
  __shared__ int   ri[32];
  __shared__ int   snxt;

  float px[8], py[8], pz[8];
#pragma unroll
  for (int k = 0; k < 8; ++k) {
    int j = tid + k * 1024;
    px[k] = p[j * 3 + 0];
    py[k] = p[j * 3 + 1];
    pz[k] = p[j * 3 + 2];
  }
  float x0 = p[0], y0 = p[1], z0 = p[2];
#pragma unroll
  for (int k = 0; k < 8; ++k) {
    int j = tid + k * 1024;
    float dx = px[k] - x0, dy = py[k] - y0, dz = pz[k] - z0;
    dsh[j] = dx * dx + dy * dy + dz * dz;
  }
  if (tid == 0) {
    centers[((size_t)b * MPTS + 0) * 3 + 0] = x0;
    centers[((size_t)b * MPTS + 0) * 3 + 1] = y0;
    centers[((size_t)b * MPTS + 0) * 3 + 2] = z0;
  }
  __syncthreads();

  for (int i = 1; i < MPTS; ++i) {
    float bv = -1.0f;
    int   bi = NPTS;
#pragma unroll
    for (int k = 0; k < 8; ++k) {
      int j = tid + k * 1024;
      float v = dsh[j];
      if (v > bv || (v == bv && j < bi)) { bv = v; bi = j; }
    }
#pragma unroll
    for (int off = 16; off > 0; off >>= 1) {
      float ov = __shfl_xor(bv, off, 32);
      int   oi = __shfl_xor(bi, off, 32);
      if (ov > bv || (ov == bv && oi < bi)) { bv = ov; bi = oi; }
    }
    if (lane == 0) { rv[wv] = bv; ri[wv] = bi; }
    __syncthreads();
    if (wv == 0) {
      bv = rv[lane];
      bi = ri[lane];
#pragma unroll
      for (int off = 16; off > 0; off >>= 1) {
        float ov = __shfl_xor(bv, off, 32);
        int   oi = __shfl_xor(bi, off, 32);
        if (ov > bv || (ov == bv && oi < bi)) { bv = ov; bi = oi; }
      }
      if (lane == 0) snxt = bi;
    }
    __syncthreads();
    int   nxt = snxt;
    float nx = p[nxt * 3 + 0], ny = p[nxt * 3 + 1], nz = p[nxt * 3 + 2];
    if (tid == 0) {
      centers[((size_t)b * MPTS + i) * 3 + 0] = nx;
      centers[((size_t)b * MPTS + i) * 3 + 1] = ny;
      centers[((size_t)b * MPTS + i) * 3 + 2] = nz;
    }
#pragma unroll
    for (int k = 0; k < 8; ++k) {
      int j = tid + k * 1024;
      float dx = px[k] - nx, dy = py[k] - ny, dz = pz[k] - nz;
      float nd = dx * dx + dy * dy + dz * dz;
      dsh[j] = fminf(dsh[j], nd);
    }
  }
}

// ---------------------------------------------------------------------------
// Kernel 2: ball query, exact lax.top_k order via 64-bit (d2bits<<32)|idx keys.
// One block (256 threads) per center; 32 d2 values per thread in registers.
// ---------------------------------------------------------------------------
__global__ void ballq_kernel(const float* __restrict__ pos,
                             const float* __restrict__ centers,
                             int* __restrict__ idxw, int* __restrict__ maskw) {
  const int cid  = blockIdx.x;
  const int b    = cid / MPTS;
  const int tid  = threadIdx.x;
  const int lane = tid & 31;
  const float* p = pos + (size_t)b * NPTS * 3;

  float c0 = centers[(size_t)cid * 3 + 0];
  float c1 = centers[(size_t)cid * 3 + 1];
  float c2 = centers[(size_t)cid * 3 + 2];

  float d2r[32];
#pragma unroll
  for (int k = 0; k < 32; ++k) {
    int j = tid + k * 256;
    float dx = p[j * 3 + 0] - c0;
    float dy = p[j * 3 + 1] - c1;
    float dz = p[j * 3 + 2] - c2;
    float d2 = dx * dx + dy * dy + dz * dz;
    d2r[k] = (d2 <= R2C) ? d2 : BIGF;
  }

  __shared__ long long rk[8];
  __shared__ long long sprev;
  if (tid == 0) sprev = -1ll;
  __syncthreads();

  for (int k = 0; k < KNN; ++k) {
    long long prev = sprev;
    long long best = 0x7fffffffffffffffll;
#pragma unroll
    for (int q = 0; q < 32; ++q) {
      int j = tid + q * 256;
      long long key = ((long long)(unsigned)__float_as_uint(d2r[q]) << 32) | (long long)j;
      if (key > prev && key < best) best = key;
    }
#pragma unroll
    for (int off = 16; off > 0; off >>= 1) {
      long long ok = __shfl_xor(best, off, 32);
      if (ok < best) best = ok;
    }
    if (lane == 0) rk[tid >> 5] = best;
    __syncthreads();
    if (tid == 0) {
      long long m = rk[0];
#pragma unroll
      for (int w = 1; w < 8; ++w)
        if (rk[w] < m) m = rk[w];
      int pi = (int)(m & 0xffffffffll);
      unsigned dbits = (unsigned)((unsigned long long)m >> 32);
      idxw[(size_t)cid * KNN + k]  = pi;
      maskw[(size_t)cid * KNN + k] = (__uint_as_float(dbits) <= R2C) ? 1 : 0;
      sprev = m;
    }
    __syncthreads();
  }
}

// ---------------------------------------------------------------------------
// Kernel 3: gather + 3-layer MLP on bf16 WMMA + fused masked max-pool.
// One block (8 waves) per center: 64 rows, K(padded)=64 per layer.
// ---------------------------------------------------------------------------
__device__ __forceinline__ bf16x16 load_frag(const __bf16* base, int rowStart,
                                             int kStart, int lane) {
  // 16-bit A/B fragment layout (CDNA5 5.7.12.2):
  // lanes 0-15: row rowStart+lane, K in {0..7} and {16..23}
  // lanes 16-31: row rowStart+lane-16, K in {8..15} and {24..31}
  int r = rowStart + (lane & 15);
  int o = ((lane >> 4) & 1) ? 8 : 0;
  const __bf16* rp = base + r * 64 + kStart;
  bf16x8 lo = *reinterpret_cast<const bf16x8*>(rp + o);
  bf16x8 hi = *reinterpret_cast<const bf16x8*>(rp + 16 + o);
  union U { bf16x8 h[2]; bf16x16 v; } u;
  u.h[0] = lo;
  u.h[1] = hi;
  return u.v;
}

template <int NCT, bool LAST>
__device__ __forceinline__ void layer_compute(const __bf16* src, __bf16* dst,
                                              const __bf16* wt, const float* bias,
                                              const int* msk, float* outv,
                                              int wv, int lane) {
  for (int tile = wv; tile < 4 * NCT; tile += 8) {
    int tm = tile / NCT;
    int tn = tile % NCT;
    f32x8 acc = {};
#pragma unroll
    for (int kc = 0; kc < 2; ++kc) {
      bf16x16 a  = load_frag(src, tm * 16, kc * 32, lane);
      bf16x16 bb = load_frag(wt,  tn * 16, kc * 32, lane);
      acc = __builtin_amdgcn_wmma_f32_16x16x32_bf16(false, a, false, bb,
                                                    (short)0, acc, false, false);
    }
    int col = tn * 16 + (lane & 15);
    int hi8 = (lane >> 4) ? 8 : 0;
    if (!LAST) {
#pragma unroll
      for (int r = 0; r < 8; ++r) {
        int row = tm * 16 + r + hi8;
        float v = acc[r] + bias[col];
        v = v > 0.f ? v : 0.f;
        dst[row * 64 + col] = (__bf16)v;
      }
    } else {
      float best = -BIGF;
#pragma unroll
      for (int r = 0; r < 8; ++r) {
        int row = tm * 16 + r + hi8;
        float v = acc[r] + bias[col];
        v = v > 0.f ? v : 0.f;
        if (!msk[row]) v = -BIGF;
        best = v > best ? v : best;
      }
      // candidates are >= 0 or exactly -BIG: int-bit atomicMax is order-correct
      atomicMax(reinterpret_cast<int*>(&outv[col]), __float_as_int(best));
    }
  }
}

__global__ void mlp_kernel(const float* __restrict__ x,
                           const float* __restrict__ pos,
                           const float* __restrict__ centers,
                           const int* __restrict__ idxw,
                           const int* __restrict__ maskw,
                           const float* __restrict__ W1, const float* __restrict__ b1,
                           const float* __restrict__ W2, const float* __restrict__ b2,
                           const float* __restrict__ W3, const float* __restrict__ b3,
                           float* __restrict__ xout, float* __restrict__ posout,
                           int* __restrict__ batchout) {
  const int cid  = blockIdx.x;
  const int b    = cid / MPTS;
  const int tid  = threadIdx.x;
  const int lane = tid & 31;
  const int wv   = tid >> 5;

  __shared__ __align__(16) __bf16 Abuf[64 * 64];
  __shared__ __align__(16) __bf16 Hbuf[64 * 64];
  __shared__ __align__(16) __bf16 Wt[128 * 64];
  __shared__ float bias_s[128];
  __shared__ float outv[128];
  __shared__ int   ids[64];
  __shared__ int   msk[64];

  if (tid < 64) {
    ids[tid] = idxw[(size_t)cid * KNN + tid];
    msk[tid] = maskw[(size_t)cid * KNN + tid];
  }
  if (tid < 128) outv[tid] = -BIGF;
  float c0 = centers[(size_t)cid * 3 + 0];
  float c1 = centers[(size_t)cid * 3 + 1];
  float c2 = centers[(size_t)cid * 3 + 2];
  __syncthreads();

  // Gather A = [x_j | rel | 0-pad] as bf16, rows = 64 neighbors, K padded to 64
  for (int e = tid; e < 64 * 64; e += 256) {
    int row = e >> 6, col = e & 63;
    int pi = ids[row];
    float v = 0.f;
    if (col < DIN)            v = x[((size_t)b * NPTS + pi) * DIN + col];
    else if (col == DIN + 0)  v = pos[((size_t)b * NPTS + pi) * 3 + 0] - c0;
    else if (col == DIN + 1)  v = pos[((size_t)b * NPTS + pi) * 3 + 1] - c1;
    else if (col == DIN + 2)  v = pos[((size_t)b * NPTS + pi) * 3 + 2] - c2;
    Abuf[e] = (__bf16)v;
  }
  // W1^T: Wt[n*64 + k], k >= 35 zero-padded
  for (int e = tid; e < 64 * 64; e += 256) {
    int n = e >> 6, kk = e & 63;
    float v = (kk < DIN + 3) ? W1[kk * 64 + n] : 0.f;
    Wt[e] = (__bf16)v;
  }
  if (tid < 64) bias_s[tid] = b1[tid];
  __syncthreads();

  layer_compute<4, false>(Abuf, Hbuf, Wt, bias_s, nullptr, nullptr, wv, lane);
  __syncthreads();

  for (int e = tid; e < 64 * 64; e += 256) {
    int n = e >> 6, kk = e & 63;
    Wt[e] = (__bf16)W2[kk * 64 + n];
  }
  if (tid < 64) bias_s[tid] = b2[tid];
  __syncthreads();

  layer_compute<4, false>(Hbuf, Abuf, Wt, bias_s, nullptr, nullptr, wv, lane);
  __syncthreads();

  for (int e = tid; e < 128 * 64; e += 256) {
    int n = e >> 6, kk = e & 63;
    Wt[e] = (__bf16)W3[kk * 128 + n];
  }
  if (tid < 128) bias_s[tid] = b3[tid];
  __syncthreads();

  layer_compute<8, true>(Abuf, nullptr, Wt, bias_s, msk, outv, wv, lane);
  __syncthreads();

  if (tid < 128) xout[(size_t)cid * 128 + tid] = outv[tid];
  if (tid == 128) {
    posout[(size_t)cid * 3 + 0] = c0;
    posout[(size_t)cid * 3 + 1] = c1;
    posout[(size_t)cid * 3 + 2] = c2;
  }
  if (tid == 129) batchout[cid] = b;
}

// ---------------------------------------------------------------------------
extern "C" void kernel_launch(void* const* d_in, const int* in_sizes, int n_in,
                              void* d_out, int out_size, void* d_ws, size_t ws_size,
                              hipStream_t stream) {
  (void)in_sizes; (void)n_in; (void)out_size; (void)ws_size;

  const float* x   = (const float*)d_in[0];
  const float* pos = (const float*)d_in[1];
  // d_in[2] = batch (int64), derivable -> unused
  const float* W1 = (const float*)d_in[3];
  const float* b1 = (const float*)d_in[4];
  const float* W2 = (const float*)d_in[5];
  const float* b2 = (const float*)d_in[6];
  const float* W3 = (const float*)d_in[7];
  const float* b3 = (const float*)d_in[8];

  float* centers = (float*)d_ws;                         // 8192*3 f32
  int*   idxw    = (int*)(centers + (size_t)BEV * MPTS * 3);   // 8192*64 i32
  int*   maskw   = idxw + (size_t)BEV * MPTS * KNN;            // 8192*64 i32

  float* o        = (float*)d_out;
  float* xout     = o;                                   // [8192,128]
  float* posout   = o + (size_t)BEV * MPTS * 128;        // [8192,3]
  int*   batchout = (int*)(posout + (size_t)BEV * MPTS * 3);   // [8192] i32

  fps_kernel<<<BEV, 1024, 0, stream>>>(pos, centers);
  ballq_kernel<<<BEV * MPTS, 256, 0, stream>>>(pos, centers, idxw, maskw);
  mlp_kernel<<<BEV * MPTS, 256, 0, stream>>>(x, pos, centers, idxw, maskw,
                                             W1, b1, W2, b2, W3, b3,
                                             xout, posout, batchout);
}
